// FouriDecoderBlock_283467842834
// MI455X (gfx1250) — matrix operand order repaired and checked by reading, not compiled, not run on hardware
//
#include <hip/hip_runtime.h>

// Problem constants (match reference)
#define CB   4
#define CS   2048
#define CD   512
#define CH   4
#define CMID 2048

// GEMM tiling: 256 threads = 8 wave32; block tile 128x128, wave tile 64x32, K-step 32
#define BM 128
#define BN 128
#define BK 32
#define APITCH 40   // LDS row pitch in bf16 halves (padded; rows stay 16B aligned)

#define F_AT   1    // A operand stored [K,M] (use A^T)
#define F_ACC  2    // C += r (fp32 C only)
#define F_NEG  4    // r = -r
#define F_BIAS 8    // r += bias[col]
#define F_SELU 16   // r = selu(r)
#define F_C16  32   // store C as bf16

typedef __attribute__((ext_vector_type(16))) __bf16 v16bf;
typedef __attribute__((ext_vector_type(8)))  float  v8f;
typedef __attribute__((ext_vector_type(4)))  float  v4f;
typedef __attribute__((ext_vector_type(4)))  unsigned v4u;

struct GemmP {
  const unsigned short* A;   // bf16
  const unsigned short* Bm;  // bf16
  void*        C;            // fp32 or bf16 (F_C16)
  const float* bias;
  long long aBatch, bBatch, cBatchDiv, cBatchMod;
  int aDiv, bMod, cDiv, cMod, biasMod;
  int M, N, K;
  int lda, ldb, ldc, cStrideN;   // element strides
};

__device__ __forceinline__ unsigned short f2bf(float f) {
  unsigned u = __float_as_uint(f);
  u += 0x7FFFu + ((u >> 16) & 1u);   // RNE fp32 -> bf16
  return (unsigned short)(u >> 16);
}

union FragU  { v4u q[2]; v16bf v; };
union Half8U { v4u u; unsigned short h[8]; };

// C[M,N] = op(A)[M,K]*B[K,N] (+bias/selu/neg/acc), batched over grid.z.
// bf16 operands staged to LDS (A as [m][k], B transposed to [n][k]) so WMMA
// fragments are contiguous 16B ds reads with A and B in the same contiguous k order.
// Double-buffered LDS + register prefetch, FLAGS compile-time (no scalar
// branching in the K-loop), last iteration peeled.
template <int FLAGS>
__global__ __launch_bounds__(256) void gemm_bf16_wmma(GemmP p) {
  __shared__ unsigned short As[2][BM * APITCH];
  __shared__ unsigned short Bs[2][BN * APITCH];

  const int t    = threadIdx.x;
  const int lane = t & 31;
  const int wave = t >> 5;
  const int g    = lane >> 4;    // half-wave group
  const int l15  = lane & 15;
  const int wm   = wave >> 2;    // 2 waves along M
  const int wn   = wave & 3;     // 4 waves along N
  const int m0   = blockIdx.y * BM;
  const int n0   = blockIdx.x * BN;
  const int z    = blockIdx.z;

  const unsigned short* Ag = p.A  + (long long)(z / p.aDiv) * p.aBatch;
  const unsigned short* Bg = p.Bm + (long long)(z % p.bMod) * p.bBatch;
  const long long cOff = (long long)(z / p.cDiv) * p.cBatchDiv
                       + (long long)(z % p.cMod) * p.cBatchMod;

  v4u ra[2], rb[2];   // prefetch registers (16 bf16 of A, 16 of B per thread)

  auto prefetch = [&](int k0) {
    if constexpr (FLAGS & F_AT) {
      #pragma unroll
      for (int c = 0; c < 2; ++c) {
        int idx = t + 256 * c;
        int kk = idx >> 4, m8 = (idx & 15) << 3;
        ra[c] = *reinterpret_cast<const v4u*>(Ag + (long long)(k0 + kk) * p.lda + (m0 + m8));
      }
    } else {
      #pragma unroll
      for (int c = 0; c < 2; ++c) {
        int idx = t + 256 * c;
        int r = idx >> 2, c8 = (idx & 3) << 3;
        ra[c] = *reinterpret_cast<const v4u*>(Ag + (long long)(m0 + r) * p.lda + (k0 + c8));
      }
    }
    #pragma unroll
    for (int c = 0; c < 2; ++c) {
      int idx = t + 256 * c;
      int kk = idx >> 4, n8 = (idx & 15) << 3;
      rb[c] = *reinterpret_cast<const v4u*>(Bg + (long long)(k0 + kk) * p.ldb + (n0 + n8));
    }
  };

  auto stage = [&](int buf) {
    if constexpr (FLAGS & F_AT) {
      #pragma unroll
      for (int c = 0; c < 2; ++c) {
        int idx = t + 256 * c;
        int kk = idx >> 4, m8 = (idx & 15) << 3;
        Half8U w; w.u = ra[c];
        #pragma unroll
        for (int j = 0; j < 8; ++j) As[buf][(m8 + j) * APITCH + kk] = w.h[j];
      }
    } else {
      #pragma unroll
      for (int c = 0; c < 2; ++c) {
        int idx = t + 256 * c;
        int r = idx >> 2, c8 = (idx & 3) << 3;
        *reinterpret_cast<v4u*>(&As[buf][r * APITCH + c8]) = ra[c];
      }
    }
    #pragma unroll
    for (int c = 0; c < 2; ++c) {
      int idx = t + 256 * c;
      int kk = idx >> 4, n8 = (idx & 15) << 3;
      Half8U w; w.u = rb[c];
      #pragma unroll
      for (int j = 0; j < 8; ++j) Bs[buf][(n8 + j) * APITCH + kk] = w.h[j];
    }
  };

  v8f zero = {};
  v8f acc[4][2];
  for (int mi = 0; mi < 4; ++mi)
    for (int ni = 0; ni < 2; ++ni) acc[mi][ni] = zero;

  auto compute = [&](int buf) {
    // B fragments: lane holds column n = l15, K = 16g..16g+15 (contiguous 32B)
    FragU bfr[2];
    #pragma unroll
    for (int ni = 0; ni < 2; ++ni) {
      const unsigned short* bp = &Bs[buf][(wn * 32 + ni * 16 + l15) * APITCH + 16 * g];
      bfr[ni].q[0] = *reinterpret_cast<const v4u*>(bp);
      bfr[ni].q[1] = *reinterpret_cast<const v4u*>(bp + 8);
    }
    // A fragments: lane row m = l15, K = 16g..16g+15 (contiguous 32B)
    #pragma unroll
    for (int mi = 0; mi < 4; ++mi) {
      FragU af;
      const unsigned short* ap = &As[buf][(wm * 64 + mi * 16 + l15) * APITCH + 16 * g];
      af.q[0] = *reinterpret_cast<const v4u*>(ap);
      af.q[1] = *reinterpret_cast<const v4u*>(ap + 8);
      #pragma unroll
      for (int ni = 0; ni < 2; ++ni) {
        acc[mi][ni] = __builtin_amdgcn_wmma_f32_16x16x32_bf16(
            false, af.v, false, bfr[ni].v, (short)0, acc[mi][ni], false, false);
      }
    }
  };

  prefetch(0);
  stage(0);
  const int T = p.K / BK;
  for (int kt = 0; kt < T - 1; ++kt) {   // steady state: no conditionals
    __syncthreads();                     // staged tile kt visible to all waves
    prefetch((kt + 1) * BK);             // overlaps with compute below
    compute(kt & 1);
    stage((kt + 1) & 1);                 // other buffer; safe after top barrier
  }
  __syncthreads();
  compute((T - 1) & 1);                  // peeled last iteration

  // Epilogue. C/D layout: element e of v8f = (row m = e + 8g, col n = l15).
  const long long biasOff = (long long)(z % p.biasMod) * p.N;
  #pragma unroll
  for (int mi = 0; mi < 4; ++mi) {
    #pragma unroll
    for (int ni = 0; ni < 2; ++ni) {
      const int col  = n0 + wn * 32 + ni * 16 + l15;
      const int row0 = m0 + wm * 64 + mi * 16 + 8 * g;
      float bb = 0.0f;
      if constexpr (FLAGS & F_BIAS) bb = p.bias[biasOff + col];
      #pragma unroll
      for (int e = 0; e < 8; ++e) {
        float r = acc[mi][ni][e] + bb;
        if constexpr (FLAGS & F_NEG) r = -r;
        if constexpr (FLAGS & F_SELU)
          r = (r > 0.0f) ? 1.0507009873554805f * r
                         : 1.7580993408473766f * (__expf(r) - 1.0f); // scale*alpha
        const long long idx = cOff + (long long)(row0 + e) * p.ldc
                            + (long long)col * p.cStrideN;
        if constexpr (FLAGS & F_C16) ((unsigned short*)p.C)[idx] = f2bf(r);
        else if constexpr (FLAGS & F_ACC) ((float*)p.C)[idx] += r;
        else ((float*)p.C)[idx] = r;
      }
    }
  }
}

// DFT matrices directly in bf16; angle reduced exactly via (p*s) mod N.
__global__ void gen_dft_kernel(unsigned short* Cs, unsigned short* Ssn,
                               unsigned short* Cd, unsigned short* Sd) {
  long long i = (long long)blockIdx.x * blockDim.x + threadIdx.x;
  if (i < (long long)CS * CS) {
    int pp = (int)(i / CS), ss = (int)(i % CS);
    int m = (pp * ss) % CS;
    float ang = 6.283185307179586f * (float)m / (float)CS;
    Cs[i]  = f2bf(cosf(ang));
    Ssn[i] = f2bf(sinf(ang));
  }
  if (i < (long long)CD * CD) {
    int pp = (int)(i / CD), dd = (int)(i % CD);
    int m = (pp * dd) % CD;
    float ang = 6.283185307179586f * (float)m / (float)CD;
    Cd[i] = f2bf(cosf(ang));
    Sd[i] = f2bf(sinf(ang));
  }
}

// one-time fp32 -> bf16 operand conversion
__global__ void f32_to_bf16_kernel(const float* x, unsigned short* y, long long n) {
  long long i = ((long long)blockIdx.x * blockDim.x + threadIdx.x) * 4;
  if (i + 3 < n) {
    v4f v = *reinterpret_cast<const v4f*>(x + i);
    y[i] = f2bf(v[0]); y[i + 1] = f2bf(v[1]); y[i + 2] = f2bf(v[2]); y[i + 3] = f2bf(v[3]);
  }
}

// softmax across the H=4 head axis of an [B,H,S,D] fp32 tensor -> bf16 output
__global__ void softmax_heads_kernel(const float* x, unsigned short* y) {
  long long t = (long long)blockIdx.x * blockDim.x + threadIdx.x;
  const long long per = (long long)CS * CD;
  if (t >= (long long)CB * per) return;
  long long base = (t / per) * ((long long)CH * per) + (t % per);
  float v0 = x[base], v1 = x[base + per], v2 = x[base + 2 * per], v3 = x[base + 3 * per];
  float mx = fmaxf(fmaxf(v0, v1), fmaxf(v2, v3));
  float e0 = __expf(v0 - mx), e1 = __expf(v1 - mx), e2 = __expf(v2 - mx), e3 = __expf(v3 - mx);
  float inv = 1.0f / (e0 + e1 + e2 + e3);
  y[base] = f2bf(e0 * inv);             y[base + per] = f2bf(e1 * inv);
  y[base + 2 * per] = f2bf(e2 * inv);   y[base + 3 * per] = f2bf(e3 * inv);
}

// out_row = LayerNorm(a_row + b_row)*g + be ; optional extra bf16 copy for GEMM use
__global__ __launch_bounds__(256) void ln_add_kernel(const float* a, const float* b,
                                                     const float* g, const float* be,
                                                     float* out, unsigned short* out16) {
  __shared__ float s1[256], s2[256];
  __shared__ float yrow[CD];
  const long long row = blockIdx.x;
  const float* ar = a + row * CD;
  const float* br = b + row * CD;
  float sum = 0.0f, sq = 0.0f;
  for (int i = threadIdx.x; i < CD; i += 256) {
    float y = ar[i] + br[i];
    yrow[i] = y; sum += y; sq += y * y;
  }
  s1[threadIdx.x] = sum; s2[threadIdx.x] = sq;
  __syncthreads();
  for (int off = 128; off > 0; off >>= 1) {
    if (threadIdx.x < off) { s1[threadIdx.x] += s1[threadIdx.x + off];
                             s2[threadIdx.x] += s2[threadIdx.x + off]; }
    __syncthreads();
  }
  const float mu = s1[0] / (float)CD;
  const float var = s2[0] / (float)CD - mu * mu;
  const float rs = rsqrtf(var + 1e-5f);
  for (int i = threadIdx.x; i < CD; i += 256) {
    float r = g[i] * (yrow[i] - mu) * rs + be[i];
    out[row * CD + i] = r;
    if (out16) out16[row * CD + i] = f2bf(r);
  }
}

static void launch_gemm(hipStream_t stream,
                        const unsigned short* A, long long aB, int aDiv, int lda,
                        const unsigned short* Bmat, long long bB, int bMod, int ldb,
                        void* C, long long cBD, int cDiv, long long cBM, int cMod,
                        int ldc, int cSN,
                        const float* bias, int biasMod,
                        int M, int N, int K, int Z, int flags) {
  GemmP p;
  p.A = A; p.Bm = Bmat; p.C = C; p.bias = bias ? bias : (const float*)A;
  p.aBatch = aB; p.bBatch = bB; p.cBatchDiv = cBD; p.cBatchMod = cBM;
  p.aDiv = aDiv < 1 ? 1 : aDiv; p.bMod = bMod < 1 ? 1 : bMod;
  p.cDiv = cDiv < 1 ? 1 : cDiv; p.cMod = cMod < 1 ? 1 : cMod;
  p.biasMod = biasMod < 1 ? 1 : biasMod;
  p.M = M; p.N = N; p.K = K;
  p.lda = lda; p.ldb = ldb; p.ldc = ldc; p.cStrideN = cSN;
  dim3 grid((unsigned)(N / BN), (unsigned)(M / BM), (unsigned)Z);
  dim3 blk(256);
  switch (flags) {
    case 0:
      gemm_bf16_wmma<0><<<grid, blk, 0, stream>>>(p); break;
    case F_C16:
      gemm_bf16_wmma<F_C16><<<grid, blk, 0, stream>>>(p); break;
    case (F_ACC | F_NEG):
      gemm_bf16_wmma<F_ACC | F_NEG><<<grid, blk, 0, stream>>>(p); break;
    case F_BIAS:
      gemm_bf16_wmma<F_BIAS><<<grid, blk, 0, stream>>>(p); break;
    case (F_BIAS | F_C16):
      gemm_bf16_wmma<F_BIAS | F_C16><<<grid, blk, 0, stream>>>(p); break;
    case (F_AT | F_C16):
      gemm_bf16_wmma<F_AT | F_C16><<<grid, blk, 0, stream>>>(p); break;
    case (F_BIAS | F_SELU | F_C16):
      gemm_bf16_wmma<F_BIAS | F_SELU | F_C16><<<grid, blk, 0, stream>>>(p); break;
    default:
      gemm_bf16_wmma<0><<<grid, blk, 0, stream>>>(p); break;
  }
}

extern "C" void kernel_launch(void* const* d_in, const int* in_sizes, int n_in,
                              void* d_out, int out_size, void* d_ws, size_t ws_size,
                              hipStream_t stream) {
  (void)in_sizes; (void)n_in; (void)out_size;
  typedef unsigned short u16;
  const float* x_enc = (const float*)d_in[0];
  const float* x_dec = (const float*)d_in[1];
  const float* Wq = (const float*)d_in[2];  const float* bq = (const float*)d_in[3];
  const float* Wk = (const float*)d_in[4];  const float* bk = (const float*)d_in[5];
  const float* Wv = (const float*)d_in[6];  const float* bv = (const float*)d_in[7];
  const float* Wo = (const float*)d_in[8];  const float* bo = (const float*)d_in[9];
  const float* ln1g = (const float*)d_in[10]; const float* ln1b = (const float*)d_in[11];
  const float* ln2g = (const float*)d_in[12]; const float* ln2b = (const float*)d_in[13];
  const float* W1 = (const float*)d_in[14]; const float* b1 = (const float*)d_in[15];
  const float* W2 = (const float*)d_in[16]; const float* b2 = (const float*)d_in[17];
  const float* ln3g = (const float*)d_in[18]; const float* ln3b = (const float*)d_in[19];
  float* out = (float*)d_out;

  const long long SxS  = (long long)CS * CS;
  const long long DxD  = (long long)CD * CD;
  const long long SD   = (long long)CS * CD;
  const long long BSD  = (long long)CB * SD;
  const long long BHSD = (long long)CB * CH * SD;
  const long long SM   = (long long)CS * CMID;
  const long long HDD  = (long long)CH * CD * CD;
  const long long DHD  = (long long)CD * CH * CD;

  // bump allocator over the workspace (256B aligned chunks)
  char* base = (char*)d_ws;
  size_t cur = 0;
  auto alloc = [&](long long bytes) -> void* {
    void* ptr = (void*)(base + cur);
    cur += (size_t)((bytes + 255) & ~255LL);
    return ptr;
  };

  u16* Csb  = (u16*)alloc(SxS * 2);    // cos DFT (seq), bf16
  u16* Ssb  = (u16*)alloc(SxS * 2);    // sin DFT (seq)
  u16* Cdb  = (u16*)alloc(DxD * 2);    // cos DFT (feat)
  u16* Sdb  = (u16*)alloc(DxD * 2);
  u16* T1b  = (u16*)alloc(BSD * 2);    // Cs*X
  u16* T2b  = (u16*)alloc(BSD * 2);    // Ss*X
  float* xf = (float*)alloc(BSD * 4);  // Re(FFT2)
  float* xd = (float*)alloc(BSD * 4);  // LN1 out (fp32, residual use)
  u16* xdb  = (u16*)alloc(BSD * 2);    // LN1 out bf16 (GEMM A)
  u16* xdecb= (u16*)alloc(BSD * 2);    // x_dec bf16
  u16* xeb  = (u16*)alloc(BSD * 2);    // x_enc bf16
  u16* Wqb  = (u16*)alloc(HDD * 2);
  u16* Wkb  = (u16*)alloc(HDD * 2);
  u16* Wvb  = (u16*)alloc(HDD * 2);
  u16* Wob  = (u16*)alloc(DHD * 2);
  u16* W1b  = (u16*)alloc((long long)CD * CMID * 2);
  u16* W2b  = (u16*)alloc((long long)CMID * CD * 2);
  float* qf = (float*)alloc(BHSD * 4); // q pre-softmax fp32
  float* kf = (float*)alloc(BHSD * 4); // k pre-softmax fp32
  u16* qb   = (u16*)alloc(BHSD * 2);   // softmax(q) bf16
  u16* kb   = (u16*)alloc(BHSD * 2);   // softmax(k) bf16
  u16* vb   = (u16*)alloc(BHSD * 2);   // v bf16
  const size_t need = cur;
  // lifetime-based aliases (disjoint lifetimes, sizes checked)
  u16*   gcb   = Csb;                  // [B,H,D,D] bf16 == S*S halves; Csb dead after G1
  u16*   outrb = kb;                   // [B,S,D*H] bf16; kb dead after gc
  u16*   midb  = vb;                   // [B,S,MID] bf16; vb dead after gc
  float* attn  = qf;                   // [B,S,D] fp32; qf dead after softmax
  float* x2    = kf;                   // [B,S,D] fp32; kf dead after softmax
  u16*   x2b   = (u16*)((char*)kf + BSD * 4);  // bf16 copy, still inside kf region
  float* ff    = xf;                   // [B,S,D] fp32; xf dead after LN1

  if (ws_size < need) return;  // workspace too small (~341 MB required)

  // 0) operand conversions (one-time; halves all repeated GEMM operand traffic)
  {
    auto conv = [&](const float* s, u16* d, long long n) {
      f32_to_bf16_kernel<<<(unsigned)((n / 4 + 255) / 256), 256, 0, stream>>>(s, d, n);
    };
    conv(x_dec, xdecb, BSD); conv(x_enc, xeb, BSD);
    conv(Wq, Wqb, HDD); conv(Wk, Wkb, HDD); conv(Wv, Wvb, HDD);
    conv(Wo, Wob, DHD); conv(W1, W1b, (long long)CD * CMID); conv(W2, W2b, (long long)CMID * CD);
  }
  gen_dft_kernel<<<(unsigned)((SxS + 255) / 256), 256, 0, stream>>>(Csb, Ssb, Cdb, Sdb);

  // 1) Fourier mixing: xf = Cs*X*Cd - Ss*X*Sd (DFT matrices symmetric)
  launch_gemm(stream, Csb, 0, 1, CS, xdecb, SD, CB, CD,
              T1b, SD, 1, 0, 1, CD, 1, nullptr, 1, CS, CD, CS, CB, F_C16);
  launch_gemm(stream, Ssb, 0, 1, CS, xdecb, SD, CB, CD,
              T2b, SD, 1, 0, 1, CD, 1, nullptr, 1, CS, CD, CS, CB, F_C16);
  launch_gemm(stream, T1b, SD, 1, CD, Cdb, 0, 1, CD,
              xf, SD, 1, 0, 1, CD, 1, nullptr, 1, CS, CD, CD, CB, 0);
  launch_gemm(stream, T2b, SD, 1, CD, Sdb, 0, 1, CD,
              xf, SD, 1, 0, 1, CD, 1, nullptr, 1, CS, CD, CD, CB, F_ACC | F_NEG);

  // 2) xd = LN(x_dec + xf)  (fp32 + bf16 copy)
  ln_add_kernel<<<CB * CS, 256, 0, stream>>>(x_dec, xf, ln1g, ln1b, xd, xdb);

  // 3) q/k/v per-head projections (Z = B*H)
  launch_gemm(stream, xdb, SD, CH, CD, Wqb, DxD, CH, CD,
              qf, SD, 1, 0, 1, CD, 1, bq, CH, CS, CD, CD, CB * CH, F_BIAS);
  launch_gemm(stream, xeb, SD, CH, CD, Wkb, DxD, CH, CD,
              kf, SD, 1, 0, 1, CD, 1, bk, CH, CS, CD, CD, CB * CH, F_BIAS);
  launch_gemm(stream, xeb, SD, CH, CD, Wvb, DxD, CH, CD,
              vb, SD, 1, 0, 1, CD, 1, bv, CH, CS, CD, CD, CB * CH, F_BIAS | F_C16);

  // 4) softmax over head axis -> bf16
  softmax_heads_kernel<<<(unsigned)((BSD + 255) / 256), 256, 0, stream>>>(qf, qb);
  softmax_heads_kernel<<<(unsigned)((BSD + 255) / 256), 256, 0, stream>>>(kf, kb);

  // 5) gc[b,h] = k[b,h]^T (DxS) * v[b,h] (SxD)
  launch_gemm(stream, kb, SD, 1, CD, vb, SD, CB * CH, CD,
              gcb, DxD, 1, 0, 1, CD, 1, nullptr, 1, CD, CD, CS, CB * CH, F_AT | F_C16);

  // 6) outr[b,s,d*H+h] = (q[b,h]*gc[b,h])[s,d]  ('b h s d -> b s (d h)' via strided store)
  launch_gemm(stream, qb, SD, 1, CD, gcb, DxD, CB * CH, CD,
              outrb, SD * CH, CH, 1, CH, CD * CH, CH,
              nullptr, 1, CS, CD, CD, CB * CH, F_C16);

  // 7) attn = outr * Wo + bo
  launch_gemm(stream, outrb, SD * CH, 1, CD * CH, Wob, 0, 1, CD,
              attn, SD, 1, 0, 1, CD, 1, bo, 1, CS, CD, CD * CH, CB, F_BIAS);

  // 8) x2 = LN(xd + attn)  (fp32 + bf16 copy)
  ln_add_kernel<<<CB * CS, 256, 0, stream>>>(xd, attn, ln2g, ln2b, x2, x2b);

  // 9) mid = selu(x2*W1 + b1)  (bf16-only output, feeds next GEMM)
  launch_gemm(stream, x2b, SD, 1, CD, W1b, 0, 1, CMID,
              midb, SM, 1, 0, 1, CMID, 1, b1, 1, CS, CMID, CD, CB, F_BIAS | F_SELU | F_C16);

  // 10) ff = mid * W2 + b2
  launch_gemm(stream, midb, SM, 1, CMID, W2b, 0, 1, CD,
              ff, SD, 1, 0, 1, CD, 1, b2, 1, CS, CD, CMID, CB, F_BIAS);

  // 11) out = LN(x2 + ff)
  ln_add_kernel<<<CB * CS, 256, 0, stream>>>(x2, ff, ln3g, ln3b, out, nullptr);
}
